// QuantizedNovaeLinear_38646115729826
// MI455X (gfx1250) — matrix operand order repaired and checked by reading, not compiled, not run on hardware
//
#include <hip/hip_runtime.h>

#define BATCH 8192
#define IN_F  1024
#define OUT_F 1024

typedef __attribute__((ext_vector_type(16))) _Float16 v16h;
typedef __attribute__((ext_vector_type(8)))  _Float16 v8h;
typedef __attribute__((ext_vector_type(4)))  _Float16 v4h;
typedef __attribute__((ext_vector_type(8)))  float    v8f;
typedef __attribute__((ext_vector_type(4)))  float    v4f;

// ---------------------------------------------------------------------------
// Kernel 1: split fp32 -> (f16 hi, f16 lo) with lo = f16(v - float(hi)).
// hi+lo carries ~22 mantissa bits -> near-fp32 when recombined via 3 WMMAs.
// ---------------------------------------------------------------------------
__global__ __launch_bounds__(256)
void split_f32_to_f16x2(const float* __restrict__ src,
                        _Float16* __restrict__ hi,
                        _Float16* __restrict__ lo,
                        int n4)
{
    int idx    = blockIdx.x * blockDim.x + threadIdx.x;
    int stride = gridDim.x * blockDim.x;
    for (int i = idx; i < n4; i += stride) {
        v4f v = ((const v4f*)src)[i];
        v4h h, l;
#pragma unroll
        for (int e = 0; e < 4; ++e) {
            _Float16 hh = (_Float16)v[e];
            _Float16 ll = (_Float16)(v[e] - (float)hh);
            h[e] = hh;
            l[e] = ll;
        }
        ((v4h*)hi)[i] = h;
        ((v4h*)lo)[i] = l;
    }
}

// ---------------------------------------------------------------------------
// Kernel 2: C = A * W^T via V_WMMA_F32_16X16X32_F16, f16 hi/lo 3-product split.
// Block = 256 threads = 8 waves (2 M x 4 N), block tile 128M x 128N.
// Wave tile 64M x 32N = 4x2 WMMA tiles; K loop in chunks of 32.
//
// Addressing: per-lane CONSTANT 32-bit byte offsets (VGPR, written once) +
// uniform base pointers advanced per chunk (SGPR s_add) -> saddr-form
// global_load_b128 with immediate offsets; no per-iteration VALU address
// math, hence no WMMA WAR-hazard v_nops.
// ---------------------------------------------------------------------------
__global__ __launch_bounds__(256)
void wmma_gemm_f16x3(const _Float16* __restrict__ Ah,
                     const _Float16* __restrict__ Al,
                     const _Float16* __restrict__ Wh,
                     const _Float16* __restrict__ Wl,
                     float* __restrict__ out)
{
    const int lane = threadIdx.x & 31;
    const int wave = threadIdx.x >> 5;
    const int r    = lane & 15;   // row (A) / col (B) within 16x16 tile
    const int g    = lane >> 4;   // lane group 0/1 (K-half selector)

    const int mBase = blockIdx.x * 128 + (wave >> 2) * 64;  // wave M origin
    const int nBase = blockIdx.y * 128 + (wave & 3) * 32;   // wave N origin

    v8f acc[4][2];
#pragma unroll
    for (int t = 0; t < 4; ++t)
#pragma unroll
        for (int u = 0; u < 2; ++u)
            acc[t][u] = (v8f){0.f, 0.f, 0.f, 0.f, 0.f, 0.f, 0.f, 0.f};

    // Constant per-lane BYTE offsets (shared by hi and lo arrays).
    unsigned aOffB[4], bOffB[2];
#pragma unroll
    for (int t = 0; t < 4; ++t)
        aOffB[t] = ((unsigned)(mBase + 16 * t + r) * IN_F + 16u * g) * 2u;
#pragma unroll
    for (int u = 0; u < 2; ++u)
        bOffB[u] = ((unsigned)(nBase + 16 * u + r) * IN_F + 16u * g) * 2u;

    // Uniform base pointers, advanced 64 bytes (32 halves) per K-chunk.
    const char* pAh = (const char*)Ah;
    const char* pAl = (const char*)Al;
    const char* pWh = (const char*)Wh;
    const char* pWl = (const char*)Wl;

#pragma unroll 4
    for (int kc = 0; kc < IN_F / 32; ++kc) {
        // --- A fragments: lane group g holds K = 16g..16g+15 (contiguous)
        v16h ah[4], al[4];
#pragma unroll
        for (int t = 0; t < 4; ++t) {
            v8h h0 = *(const v8h*)(pAh + aOffB[t]);
            v8h h1 = *(const v8h*)(pAh + aOffB[t] + 16);
            v8h l0 = *(const v8h*)(pAl + aOffB[t]);
            v8h l1 = *(const v8h*)(pAl + aOffB[t] + 16);
            ah[t] = __builtin_shufflevector(h0, h1, 0, 1, 2, 3, 4, 5, 6, 7,
                                            8, 9, 10, 11, 12, 13, 14, 15);
            al[t] = __builtin_shufflevector(l0, l1, 0, 1, 2, 3, 4, 5, 6, 7,
                                            8, 9, 10, 11, 12, 13, 14, 15);
        }
        // --- B fragments: lane group g holds K = 16g..16g+15 (contiguous)
        v16h bh[2], bl[2];
#pragma unroll
        for (int u = 0; u < 2; ++u) {
            bh[u] = *(const v16h*)(pWh + bOffB[u]);
            bl[u] = *(const v16h*)(pWl + bOffB[u]);
        }
        // --- 3-product split: hh + hl + lh  (ll term ~2^-22, dropped)
#pragma unroll
        for (int t = 0; t < 4; ++t) {
#pragma unroll
            for (int u = 0; u < 2; ++u) {
                acc[t][u] = __builtin_amdgcn_wmma_f32_16x16x32_f16(
                    false, ah[t], false, bh[u], (short)0, acc[t][u], false, false);
                acc[t][u] = __builtin_amdgcn_wmma_f32_16x16x32_f16(
                    false, ah[t], false, bl[u], (short)0, acc[t][u], false, false);
                acc[t][u] = __builtin_amdgcn_wmma_f32_16x16x32_f16(
                    false, al[t], false, bh[u], (short)0, acc[t][u], false, false);
            }
        }
        pAh += 64; pAl += 64; pWh += 64; pWl += 64;  // uniform SALU bumps
    }

    // C/D layout: VGPR e -> M = 16t + e + 8g, N = lane&15.
#pragma unroll
    for (int t = 0; t < 4; ++t) {
#pragma unroll
        for (int u = 0; u < 2; ++u) {
            unsigned o = (unsigned)(mBase + 16 * t + 8 * g) * OUT_F
                       + (unsigned)(nBase + 16 * u + r);
#pragma unroll
            for (int e = 0; e < 8; ++e)
                out[o + (unsigned)e * OUT_F] = acc[t][u][e];
        }
    }
}

// ---------------------------------------------------------------------------
extern "C" void kernel_launch(void* const* d_in, const int* in_sizes, int n_in,
                              void* d_out, int out_size, void* d_ws, size_t ws_size,
                              hipStream_t stream)
{
    (void)in_sizes; (void)n_in; (void)out_size; (void)ws_size;

    const float* x = (const float*)d_in[0];
    const float* w = (const float*)d_in[1];
    // d_in[2] (bias) cancels in the STE value path: out == x @ w^T exactly.

    _Float16* xh = (_Float16*)d_ws;
    _Float16* xl = xh + (size_t)BATCH * IN_F;
    _Float16* wh = xl + (size_t)BATCH * IN_F;
    _Float16* wl = wh + (size_t)OUT_F * IN_F;
    float* out = (float*)d_out;

    {
        int n4 = (BATCH * IN_F) / 4;
        int blocks = (n4 + 255) / 256;
        if (blocks > 4096) blocks = 4096;
        split_f32_to_f16x2<<<blocks, 256, 0, stream>>>(x, xh, xl, n4);
    }
    {
        int n4 = (OUT_F * IN_F) / 4;
        int blocks = (n4 + 255) / 256;
        if (blocks > 4096) blocks = 4096;
        split_f32_to_f16x2<<<blocks, 256, 0, stream>>>(w, wh, wl, n4);
    }

    dim3 grid(BATCH / 128, OUT_F / 128);  // 64 x 8 workgroups
    wmma_gemm_f16x3<<<grid, dim3(256), 0, stream>>>(xh, xl, wh, wl, out);
}
